// LinearAttention_27496380629323
// MI455X (gfx1250) — matrix operand order repaired and checked by reading, not compiled, not run on hardware
//
#include <hip/hip_runtime.h>
#include <hip/hip_bf16.h>
#include <hip/hip_fp16.h>

// ---------------------------------------------------------------------------
// Problem constants (match reference)
// ---------------------------------------------------------------------------
#define BB   2
#define TT   8192
#define DD   1024
#define HH   8
#define DKq  128
#define DVq  128
#define CC   64
#define NCH  (TT / CC)          // 128 chunks
#define MT   (BB * TT)          // 16384 rows
#define MH   (MT * HH)          // 131072 per-head rows

typedef __attribute__((ext_vector_type(16))) _Float16 v16h;
typedef __attribute__((ext_vector_type(8)))  _Float16 v8h;
typedef __attribute__((ext_vector_type(8)))  float    v8f;
typedef __attribute__((ext_vector_type(4)))  int      v4i;

// ---------------------------------------------------------------------------
// Async global->LDS copy (gfx1250 GLOBAL_LOAD_ASYNC_TO_LDS_B128), guarded.
// Probe-learned signature: (v4i* gsrc, v4i* ldst, imm offset, imm cpol).
// ---------------------------------------------------------------------------
#if defined(__has_builtin)
#if __has_builtin(__builtin_amdgcn_global_load_async_to_lds_b128)
#define HAVE_ASYNC_LDS 1
#endif
#endif

__device__ __forceinline__ void async_cp16(const void* g, void* l) {
#ifdef HAVE_ASYNC_LDS
  __builtin_amdgcn_global_load_async_to_lds_b128((v4i*)g, (v4i*)l, 0, 0);
#else
  *reinterpret_cast<v8h*>(l) = *reinterpret_cast<const v8h*>(g);
#endif
}

__device__ __forceinline__ void wait_async0() {
#ifdef HAVE_ASYNC_LDS
#if __has_builtin(__builtin_amdgcn_s_wait_asynccnt)
  __builtin_amdgcn_s_wait_asynccnt(0);
#else
  asm volatile("s_wait_asynccnt 0x0" ::: "memory");
#endif
#endif
}

// ---------------------------------------------------------------------------
// WMMA helper (wave32): D = A(16x32 f16) * B(32x16 f16) + C(16x16 f32)
// ---------------------------------------------------------------------------
__device__ __forceinline__ v8f wmma_f16(v16h a, v16h b, v8f c) {
  return __builtin_amdgcn_wmma_f32_16x16x32_f16(false, a, false, b,
                                                (short)0, c, false, false);
}

// A fragment 16x32 from row-major [M x K] LDS tile: two 16B spans per lane.
__device__ __forceinline__ v16h frag_a_vec(const _Float16* p, int ld) {
  const int lane = threadIdx.x & 31;
  const _Float16* q = p + (lane & 15) * ld + ((lane >> 4) << 3);
  v8h lo = *reinterpret_cast<const v8h*>(q);
  v8h hi = *reinterpret_cast<const v8h*>(q + 16);
  return __builtin_shufflevector(lo, hi, 0, 1, 2, 3, 4, 5, 6, 7,
                                 8, 9, 10, 11, 12, 13, 14, 15);
}

// B fragment 32x16 where logical B[k][n] = src[n][k] (src N-major [N x K]):
// two 16B spans per lane.
__device__ __forceinline__ v16h frag_bt_vec(const _Float16* p, int ld) {
  const int lane = threadIdx.x & 31;
  const _Float16* q = p + (lane & 15) * ld + ((lane >> 4) << 3);
  v8h lo = *reinterpret_cast<const v8h*>(q);
  v8h hi = *reinterpret_cast<const v8h*>(q + 16);
  return __builtin_shufflevector(lo, hi, 0, 1, 2, 3, 4, 5, 6, 7,
                                 8, 9, 10, 11, 12, 13, 14, 15);
}

// ---------------------------------------------------------------------------
// f32 -> f16 conversion (8 elems/thread, packed v8h store)
// ---------------------------------------------------------------------------
__global__ __launch_bounds__(256) void f32_to_f16_kernel(
    const float* __restrict__ src, _Float16* __restrict__ dst, long long n) {
  long long i = ((long long)blockIdx.x * 256 + threadIdx.x) * 8;
  if (i + 7 < n) {
    float4 a = *reinterpret_cast<const float4*>(src + i);
    float4 b = *reinterpret_cast<const float4*>(src + i + 4);
    v8h p;
    p[0] = (_Float16)a.x; p[1] = (_Float16)a.y;
    p[2] = (_Float16)a.z; p[3] = (_Float16)a.w;
    p[4] = (_Float16)b.x; p[5] = (_Float16)b.y;
    p[6] = (_Float16)b.z; p[7] = (_Float16)b.w;
    *reinterpret_cast<v8h*>(dst + i) = p;
  }
}

// ---------------------------------------------------------------------------
// f32 [R x C] -> f16 transposed [C x R]
// ---------------------------------------------------------------------------
__global__ __launch_bounds__(256) void transpose_to_f16_kernel(
    const float* __restrict__ src, _Float16* __restrict__ dst, int R, int Cn) {
  long long i = (long long)blockIdx.x * 256 + threadIdx.x;
  if (i < (long long)R * Cn) {
    const int r = (int)(i / Cn), c = (int)(i % Cn);
    dst[(long long)c * R + r] = (_Float16)src[i];
  }
}

// ---------------------------------------------------------------------------
// Hadamard feature map: out = ((a + ba[d]) * (b + bb[d])) * scale (8/thread)
// ---------------------------------------------------------------------------
__global__ __launch_bounds__(256) void hadamard_fmap_kernel(
    const float* __restrict__ a, const float* __restrict__ b,
    const float* __restrict__ ba, const float* __restrict__ bb,
    float scale, _Float16* __restrict__ out, long long n) {
  long long i = ((long long)blockIdx.x * 256 + threadIdx.x) * 8;
  if (i + 7 < n) {
    const int d0 = (int)(i & (DKq - 1));   // 8-aligned within a 128 row
    v8h p;
#pragma unroll
    for (int e = 0; e < 8; ++e)
      p[e] = (_Float16)((a[i + e] + ba[d0 + e]) * (b[i + e] + bb[d0 + e]) * scale);
    *reinterpret_cast<v8h*>(out + i) = p;
  }
}

// ---------------------------------------------------------------------------
// Tiled WMMA GEMM: C[M,N](f32) = A[M,K](f16,row) * Bt[N,K](f16, pre-transposed)
// Block tile 128x128, BK=32, 8 waves, each wave -> 32x64 (2x4 wmma tiles).
// Double-buffered LDS tiles fed by async global->LDS copies.
// ---------------------------------------------------------------------------
__global__ __launch_bounds__(256) void gemm_f16_wmma_kernel(
    const _Float16* __restrict__ A, const _Float16* __restrict__ Bt,
    float* __restrict__ C, int M, int N, int K) {
  constexpr int ALD = 40;    // 32 + 8 halfs pad
  constexpr int BLD = 40;
  __shared__ alignas(16) _Float16 sA [2][128 * ALD];
  __shared__ alignas(16) _Float16 sBt[2][128 * BLD];

  const int tid  = threadIdx.x;
  const int lane = tid & 31;
  const int w    = tid >> 5;
  const int wm   = w & 3;        // 4 row groups of 32
  const int wn   = w >> 2;       // 2 col groups of 64
  const long long bm = (long long)blockIdx.y * 128;
  const long long bn = (long long)blockIdx.x * 128;

  v8f acc[2][4];
#pragma unroll
  for (int i = 0; i < 2; ++i)
#pragma unroll
    for (int j = 0; j < 4; ++j) acc[i][j] = {};

  const int arow = tid >> 1, acol = (tid & 1) * 16;  // 128 rows x 32 halfs
  const _Float16* gA0 = A  + (bm + arow) * (long long)K + acol;
  const _Float16* gB0 = Bt + (bn + arow) * (long long)K + acol;

  auto stage = [&](int k0, int buf) {
    async_cp16(gA0 + k0,     &sA [buf][arow * ALD + acol]);
    async_cp16(gA0 + k0 + 8, &sA [buf][arow * ALD + acol + 8]);
    async_cp16(gB0 + k0,     &sBt[buf][arow * BLD + acol]);
    async_cp16(gB0 + k0 + 8, &sBt[buf][arow * BLD + acol + 8]);
  };

  const int nk = K / 32;
  stage(0, 0);
  for (int kb = 0; kb < nk; ++kb) {
    wait_async0();
    __syncthreads();                      // buffer kb&1 ready for everyone
    if (kb + 1 < nk) stage((kb + 1) * 32, (kb + 1) & 1);  // overlap next fetch

    const _Float16* cA = sA [kb & 1];
    const _Float16* cB = sBt[kb & 1];
    v16h afr[2], bfr[4];
#pragma unroll
    for (int i = 0; i < 2; ++i)
      afr[i] = frag_a_vec(cA + (wm * 32 + i * 16) * ALD, ALD);
#pragma unroll
    for (int j = 0; j < 4; ++j)
      bfr[j] = frag_bt_vec(cB + (wn * 64 + j * 16) * BLD, BLD);
#pragma unroll
    for (int i = 0; i < 2; ++i)
#pragma unroll
      for (int j = 0; j < 4; ++j)
        acc[i][j] = wmma_f16(afr[i], bfr[j], acc[i][j]);
  }

  const int hi = (lane >> 4) * 8;
  const int nn = lane & 15;
#pragma unroll
  for (int i = 0; i < 2; ++i)
#pragma unroll
    for (int j = 0; j < 4; ++j)
#pragma unroll
      for (int r = 0; r < 8; ++r) {
        const long long m = bm + wm * 32 + i * 16 + r + hi;
        const long long n = bn + wn * 64 + j * 16 + nn;
        C[m * N + n] = acc[i][j][r];
      }
}

// ---------------------------------------------------------------------------
// Chunked causal linear attention. One block per (b,h), serial over chunks.
// S state in per-wave f32 WMMA accumulators (wave w owns dk rows [16w,16w+16));
// transposed f16 shadow S^T[dv][dk] in LDS feeds q@S. v^T / k^T staged
// transposed so every WMMA B operand is a contiguous ds_load_b128.
// q chunk rows arrive via async global->LDS.
// ---------------------------------------------------------------------------
__global__ __launch_bounds__(256) void linattn_chunk_kernel(
    const _Float16* __restrict__ Q, const _Float16* __restrict__ Kf,
    const _Float16* __restrict__ V, const float* __restrict__ g,
    _Float16* __restrict__ O) {
  constexpr int QLD  = 136;   // q,k row-major chunk rows (128 + 8)
  constexpr int TLD  = 72;    // k^T, v^T rows (64 + 8)
  constexpr int SLD  = 136;   // S^T shadow rows (128 + 8)
  constexpr int ALD  = 72;    // attn rows (64 + 8)
  constexpr int OLD  = 132;   // f32 o rows

  extern __shared__ char smem_raw[];
  float*    sO   = reinterpret_cast<float*>(smem_raw);   // 64*OLD f32
  float*    sDen = sO + 64 * OLD;                        // 64
  float*    sZ   = sDen + 64;                            // 128
  _Float16* sQ   = reinterpret_cast<_Float16*>(sZ + 128);
  _Float16* sK   = sQ  + 64 * QLD;                       // k row-major
  _Float16* sKt  = sK  + 64 * QLD;                       // k^T [dk][t]
  _Float16* sVt  = sKt + 128 * TLD;                      // v^T [dv][t]
  _Float16* sSt  = sVt + 128 * TLD;                      // S^T [dv][dk]
  _Float16* sA   = sSt + 128 * SLD;                      // attn [t][t']

  const int tid  = threadIdx.x;
  const int lane = tid & 31;
  const int w    = tid >> 5;
  const int b    = blockIdx.x / HH;
  const int h    = blockIdx.x % HH;
  const int hi   = (lane >> 4) * 8;
  const int nn   = lane & 15;

  v8f sAcc[8];
#pragma unroll
  for (int j = 0; j < 8; ++j) sAcc[j] = {};
  for (int i = tid; i < 128 * SLD; i += 256) sSt[i] = (_Float16)0.f;
  if (tid < 128) sZ[tid] = 0.f;
  __syncthreads();

  const long long head_off = ((long long)b * TT) * (HH * DKq) + (long long)h * DKq;
  const int row = tid >> 2;              // chunk row 0..63
  const int col = (tid & 3) * 32;        // 32 halfs per thread per buffer

  for (int ci = 0; ci < NCH; ++ci) {
    const int t0 = ci * CC;

    // ---- stage: q async row-major; k row-major and k^T,v^T scatter from registers ----
    {
      const long long gidx = head_off + (long long)(t0 + row) * (HH * DKq) + col;
      const _Float16* gq = Q + gidx;
#pragma unroll
      for (int u = 0; u < 4; ++u)
        async_cp16(gq + u * 8, sQ + row * QLD + col + u * 8);

      v8h kv[4], vv[4];
      const v8h* gk = reinterpret_cast<const v8h*>(Kf + gidx);
      const v8h* gv = reinterpret_cast<const v8h*>(V + gidx);
#pragma unroll
      for (int u = 0; u < 4; ++u) { kv[u] = gk[u]; vv[u] = gv[u]; }
      v8h* lk = reinterpret_cast<v8h*>(sK + row * QLD + col);
#pragma unroll
      for (int u = 0; u < 4; ++u) lk[u] = kv[u];
#pragma unroll
      for (int u = 0; u < 4; ++u)
#pragma unroll
        for (int e = 0; e < 8; ++e) {
          const int d = col + u * 8 + e;
          sKt[d * TLD + row] = kv[u][e];
          sVt[d * TLD + row] = vv[u][e];
        }
    }
    wait_async0();
    __syncthreads();   // B0

    // ---- attn = q k^T (64x64), causal mask, store f16 ----
    {
      const int tm = w >> 1;
#pragma unroll
      for (int jn = 0; jn < 2; ++jn) {
        const int tn = (w & 1) * 2 + jn;
        v8f c = {};
#pragma unroll
        for (int ks = 0; ks < 4; ++ks) {
          v16h a  = frag_a_vec (sQ + tm * 16 * QLD + ks * 32, QLD);
          v16h bt = frag_bt_vec(sK + tn * 16 * QLD + ks * 32, QLD);
          c = wmma_f16(a, bt, c);
        }
#pragma unroll
        for (int r = 0; r < 8; ++r) {
          const int m = tm * 16 + r + hi;
          const int n = tn * 16 + nn;
          sA[m * ALD + n] = (_Float16)((n <= m) ? c[r] : 0.0f);
        }
      }
    }
    __syncthreads();   // B1

    // ---- denominator: intra rowsum + q . Z (exclusive), vector reads ----
    if (tid < 64) {
      float s = 0.f;
      const v8h* ar = reinterpret_cast<const v8h*>(sA + tid * ALD);
#pragma unroll
      for (int u = 0; u < 8; ++u) {
        v8h t = ar[u];
#pragma unroll
        for (int e = 0; e < 8; ++e) s += (float)t[e];
      }
      const v8h* qr = reinterpret_cast<const v8h*>(sQ + tid * QLD);
#pragma unroll
      for (int u = 0; u < 16; ++u) {
        v8h t = qr[u];
#pragma unroll
        for (int e = 0; e < 8; ++e) s += (float)t[e] * sZ[u * 8 + e];
      }
      sDen[tid] = s;
    }

    // ---- o = attn @ v + q @ S  (wave: 16 t-rows x 64 dv-cols) ----
    {
      const int tm = w & 3;
      const int vn = (w >> 2) * 64;
      v8f c[4];
#pragma unroll
      for (int j = 0; j < 4; ++j) c[j] = {};
#pragma unroll
      for (int ks = 0; ks < 2; ++ks) {              // attn @ v, K = 64
        v16h a = frag_a_vec(sA + tm * 16 * ALD + ks * 32, ALD);
#pragma unroll
        for (int j = 0; j < 4; ++j) {
          v16h bb = frag_bt_vec(sVt + (vn + j * 16) * TLD + ks * 32, TLD);
          c[j] = wmma_f16(a, bb, c[j]);
        }
      }
#pragma unroll
      for (int ks = 0; ks < 4; ++ks) {              // q @ S, K = 128
        v16h a = frag_a_vec(sQ + tm * 16 * QLD + ks * 32, QLD);
#pragma unroll
        for (int j = 0; j < 4; ++j) {
          v16h bb = frag_bt_vec(sSt + (vn + j * 16) * SLD + ks * 32, SLD);
          c[j] = wmma_f16(a, bb, c[j]);
        }
      }
#pragma unroll
      for (int j = 0; j < 4; ++j)
#pragma unroll
        for (int r = 0; r < 8; ++r)
          sO[(tm * 16 + r + hi) * OLD + vn + j * 16 + nn] = c[j][r];
    }
    __syncthreads();   // B2

    // ---- epilogue: /(den+eps), RMSNorm(head), packed f16 store ----
    if (tid < 64) {
      const float inv = 1.0f / (sDen[tid] + 1e-6f);
      float ss = 0.f;
      for (int d = 0; d < DVq; ++d) {
        const float v = sO[tid * OLD + d] * inv;
        sO[tid * OLD + d] = v;
        ss += v * v;
      }
      const float rn = rsqrtf(ss * (1.0f / DVq) + 1e-5f);
      _Float16* op = O + ((long long)b * TT + (t0 + tid)) * (HH * DVq)
                       + (long long)h * DVq;
#pragma unroll
      for (int seg = 0; seg < DVq / 8; ++seg) {
        v8h pack;
#pragma unroll
        for (int e = 0; e < 8; ++e) {
          const int d = seg * 8 + e;
          pack[e] = (_Float16)(sO[tid * OLD + d] * rn * g[d]);
        }
        *reinterpret_cast<v8h*>(op + seg * 8) = pack;
      }
    }

    // ---- state update: S += k^T v (WMMA), refresh transposed f16 shadow ----
    {
#pragma unroll
      for (int ks = 0; ks < 2; ++ks) {
        v16h a = frag_a_vec(sKt + w * 16 * TLD + ks * 32, TLD);  // k^T slice
#pragma unroll
        for (int j = 0; j < 8; ++j) {
          v16h bb = frag_bt_vec(sVt + j * 16 * TLD + ks * 32, TLD);
          sAcc[j] = wmma_f16(a, bb, sAcc[j]);
        }
      }
#pragma unroll
      for (int j = 0; j < 8; ++j)
#pragma unroll
        for (int r = 0; r < 8; ++r)
          sSt[(j * 16 + nn) * SLD + w * 16 + r + hi] = (_Float16)sAcc[j][r];
    }
    // ---- Z += column-sums of k (read k^T rows, vector) ----
    if (tid < 128) {
      float z = 0.f;
      const v8h* kr = reinterpret_cast<const v8h*>(sKt + tid * TLD);
#pragma unroll
      for (int u = 0; u < 8; ++u) {
        v8h t = kr[u];
#pragma unroll
        for (int e = 0; e < 8; ++e) z += (float)t[e];
      }
      sZ[tid] += z;
    }
    __syncthreads();   // B3 (protect staging buffers before next chunk)
  }
}

// ---------------------------------------------------------------------------
// Host-side launch sequence
// ---------------------------------------------------------------------------
static inline void conv16(const float* s, _Float16* d, long long n, hipStream_t st) {
  const long long blocks = (n / 8 + 255) / 256;
  f32_to_f16_kernel<<<(unsigned)blocks, 256, 0, st>>>(s, d, n);
}
static inline void transp16(const float* s, _Float16* d, int R, int Cn, hipStream_t st) {
  const long long n = (long long)R * Cn;
  transpose_to_f16_kernel<<<(unsigned)((n + 255) / 256), 256, 0, st>>>(s, d, R, Cn);
}

extern "C" void kernel_launch(void* const* d_in, const int* in_sizes, int n_in,
                              void* d_out, int out_size, void* d_ws, size_t ws_size,
                              hipStream_t stream) {
  const float* x     = (const float*)d_in[0];
  const float* Wq    = (const float*)d_in[1];
  const float* Wk    = (const float*)d_in[2];
  const float* Wv    = (const float*)d_in[3];
  const float* Wo    = (const float*)d_in[4];
  const float* fq1w  = (const float*)d_in[5];
  const float* fq1b  = (const float*)d_in[6];
  const float* fq2w  = (const float*)d_in[7];
  const float* fq2b  = (const float*)d_in[8];
  const float* fk1w  = (const float*)d_in[9];
  const float* fk1b  = (const float*)d_in[10];
  const float* fk2w  = (const float*)d_in[11];
  const float* fk2b  = (const float*)d_in[12];
  const float* gnorm = (const float*)d_in[13];

  char* ws = (char*)d_ws;
  size_t off = 0;
  auto alloc = [&](size_t bytes) -> char* {
    char* p = ws + off;
    off += (bytes + 255) & ~(size_t)255;
    return p;
  };

  const long long nXD = (long long)MT * DD;       // 16.7M elems
  _Float16* xh   = (_Float16*)alloc(nXD * 2);
  _Float16* wqt  = (_Float16*)alloc((size_t)DD * DD * 2);   // [N,K] transposed
  _Float16* wkt  = (_Float16*)alloc((size_t)DD * DD * 2);
  _Float16* wvt  = (_Float16*)alloc((size_t)DD * DD * 2);
  _Float16* wot  = (_Float16*)alloc((size_t)DD * DD * 2);
  _Float16* f1qt = (_Float16*)alloc((size_t)DKq * DKq * 2);
  _Float16* f2qt = (_Float16*)alloc((size_t)DKq * DKq * 2);
  _Float16* f1kt = (_Float16*)alloc((size_t)DKq * DKq * 2);
  _Float16* f2kt = (_Float16*)alloc((size_t)DKq * DKq * 2);
  _Float16* qh   = (_Float16*)alloc(nXD * 2);
  _Float16* kh   = (_Float16*)alloc(nXD * 2);
  _Float16* vh   = (_Float16*)alloc(nXD * 2);
  _Float16* fqh  = (_Float16*)alloc(nXD * 2);
  _Float16* fkh  = (_Float16*)alloc(nXD * 2);
  _Float16* oh   = (_Float16*)alloc(nXD * 2);
  float*    bufA = (float*)alloc(nXD * 4);
  float*    bufB = (float*)alloc(nXD * 4);

  // 1) convert x; transpose-convert all weights to [N,K] f16
  conv16(x, xh, nXD, stream);
  transp16(Wq,   wqt,  DD,  DD,  stream);
  transp16(Wk,   wkt,  DD,  DD,  stream);
  transp16(Wv,   wvt,  DD,  DD,  stream);
  transp16(Wo,   wot,  DD,  DD,  stream);
  transp16(fq1w, f1qt, DKq, DKq, stream);
  transp16(fq2w, f2qt, DKq, DKq, stream);
  transp16(fk1w, f1kt, DKq, DKq, stream);
  transp16(fk2w, f2kt, DKq, DKq, stream);

  const dim3 blk(256);
  const dim3 gProj(DD / 128, MT / 128);       // (8, 128)
  const dim3 gHead(DKq / 128, MH / 128);      // (1, 1024)

  // 2) QKV projections (WMMA GEMM) + f16 conversion
  gemm_f16_wmma_kernel<<<gProj, blk, 0, stream>>>(xh, wqt, bufA, MT, DD, DD);
  conv16(bufA, qh, nXD, stream);
  gemm_f16_wmma_kernel<<<gProj, blk, 0, stream>>>(xh, wkt, bufA, MT, DD, DD);
  conv16(bufA, kh, nXD, stream);
  gemm_f16_wmma_kernel<<<gProj, blk, 0, stream>>>(xh, wvt, bufA, MT, DD, DD);
  conv16(bufA, vh, nXD, stream);

  // 3) Hadamard feature maps (per-head GEMMs viewed as [MH x DK])
  const unsigned nPt = (unsigned)((nXD / 8 + 255) / 256);
  const float qscale = rsqrtf((float)DKq);    // 1/sqrt(128)
  gemm_f16_wmma_kernel<<<gHead, blk, 0, stream>>>(qh, f1qt, bufA, MH, DKq, DKq);
  gemm_f16_wmma_kernel<<<gHead, blk, 0, stream>>>(qh, f2qt, bufB, MH, DKq, DKq);
  hadamard_fmap_kernel<<<nPt, blk, 0, stream>>>(bufA, bufB, fq1b, fq2b, qscale, fqh, nXD);
  gemm_f16_wmma_kernel<<<gHead, blk, 0, stream>>>(kh, f1kt, bufA, MH, DKq, DKq);
  gemm_f16_wmma_kernel<<<gHead, blk, 0, stream>>>(kh, f2kt, bufB, MH, DKq, DKq);
  hadamard_fmap_kernel<<<nPt, blk, 0, stream>>>(bufA, bufB, fk1b, fk2b, 1.0f, fkh, nXD);

  // 4) chunked causal linear attention + norm epilogue
  constexpr size_t SMEM =
      (64 * 132 + 64 + 128) * sizeof(float) +
      (2 * 64 * 136 + 2 * 128 * 72 + 128 * 136 + 64 * 72) * sizeof(_Float16);
  static_assert(SMEM < 320 * 1024, "fits WGP LDS");
  (void)hipFuncSetAttribute((const void*)linattn_chunk_kernel,
                            hipFuncAttributeMaxDynamicSharedMemorySize,
                            (int)SMEM);
  linattn_chunk_kernel<<<BB * HH, blk, SMEM, stream>>>(fqh, fkh, vh, gnorm, oh);

  // 5) output projection straight into d_out (f32)
  gemm_f16_wmma_kernel<<<gProj, blk, 0, stream>>>(oh, wot, (float*)d_out, MT, DD, DD);
}
